// TextEncoder_80616536146744
// MI455X (gfx1250) — matrix operand order repaired and checked by reading, not compiled, not run on hardware
//
#include <hip/hip_runtime.h>
#include <hip/hip_bf16.h>
#include <math.h>

typedef __attribute__((ext_vector_type(16))) __bf16 v16bf;
typedef __attribute__((ext_vector_type(8)))  __bf16 v8bf;
typedef __attribute__((ext_vector_type(8)))  float  v8f;

#define D_MODEL 1024
#define NHEAD   16
#define HDK     64
#define NLAYER  12
#define LSEQ    128
#define BATCH   32
#define NTOK    (BATCH * LSEQ)   // 4096
#define DFF     4096
#define EMBD    512

// LDS staging geometry for the GEMM A tile: 32 rows x 32 k-elems (bf16),
// row pitch 80B (16B aligned for ds_load_b128; 20-bank stride -> 16 rows hit
// 16 distinct bank groups, conflict-free fragment reads). Double buffered.
#define A_PITCH      80
#define A_TILE_BYTES (32 * A_PITCH)   // 2560B per buffer

// ---------------------------------------------------------------------------
// CDNA5 async global->LDS copy (ASYNCcnt-tracked). LDS destination address is
// a 32-bit byte offset in a VGPR; ISA 10.2: low 32 bits of a generic shared
// pointer are the LDS offset.
// ---------------------------------------------------------------------------
__device__ __forceinline__ void async_copy_b64(unsigned lds_off, const void* gaddr)
{
    asm volatile("global_load_async_to_lds_b64 %0, %1, off"
                 :: "v"(lds_off), "v"(gaddr) : "memory");
}
__device__ __forceinline__ void wait_async0()
{
    asm volatile("s_wait_asynccnt 0x0" ::: "memory");
}

// ---------------------------------------------------------------------------
// Embedding + sinusoidal positional encoding
// ---------------------------------------------------------------------------
__global__ __launch_bounds__(256)
void embed_kernel(const int* __restrict__ text, const float* __restrict__ emb,
                  const float* __restrict__ cls, float* __restrict__ x)
{
    const int row = blockIdx.x;          // 0..NTOK-1
    const int b = row >> 7;
    const int s = row & (LSEQ - 1);
    const float* src = (s == 0) ? cls
                                : emb + (size_t)text[b * (LSEQ - 1) + (s - 1)] * D_MODEL;
    for (int i = threadIdx.x; i < D_MODEL; i += 256) {
        float fi  = (float)(i & ~1);
        float div = powf(10000.0f, fi * (1.0f / (float)D_MODEL));
        float ang = (float)s / div;
        float pe  = (i & 1) ? cosf(ang) : sinf(ang);
        x[(size_t)row * D_MODEL + i] = src[i] + pe;
    }
}

// ---------------------------------------------------------------------------
// LayerNorm (population variance, eps=1e-5) -> bf16 activations for WMMA
// ---------------------------------------------------------------------------
__global__ __launch_bounds__(256)
void layernorm_kernel(const float* __restrict__ x, const float* __restrict__ g,
                      const float* __restrict__ bta, __bf16* __restrict__ out)
{
    const int row = blockIdx.x;
    __shared__ float s1[256], s2[256];
    const float* xr = x + (size_t)row * D_MODEL;
    float a0 = 0.f, a1 = 0.f;
    for (int i = threadIdx.x; i < D_MODEL; i += 256) {
        float t = xr[i]; a0 += t; a1 += t * t;
    }
    s1[threadIdx.x] = a0; s2[threadIdx.x] = a1; __syncthreads();
    for (int off = 128; off > 0; off >>= 1) {
        if (threadIdx.x < off) {
            s1[threadIdx.x] += s1[threadIdx.x + off];
            s2[threadIdx.x] += s2[threadIdx.x + off];
        }
        __syncthreads();
    }
    const float mean = s1[0] * (1.0f / D_MODEL);
    const float var  = s2[0] * (1.0f / D_MODEL) - mean * mean;
    const float inv  = rsqrtf(var + 1e-5f);
    for (int i = threadIdx.x; i < D_MODEL; i += 256)
        out[(size_t)row * D_MODEL + i] = (__bf16)((xr[i] - mean) * inv * g[i] + bta[i]);
}

// ---------------------------------------------------------------------------
// Weight packing: fp32 [K,N] row-major -> bf16 N-major Bt[n*K + k]
// ---------------------------------------------------------------------------
__global__ __launch_bounds__(256)
void pack_w_kernel(const float* __restrict__ W, __bf16* __restrict__ Bt, int K, int N)
{
    const size_t total = (size_t)K * N;
    for (size_t i = (size_t)blockIdx.x * blockDim.x + threadIdx.x; i < total;
         i += (size_t)gridDim.x * blockDim.x) {
        size_t n = i / (size_t)K, k = i % (size_t)K;
        Bt[i] = (__bf16)W[k * (size_t)N + n];
    }
}

// Wq/Wk/Wv per layer: fp32 [H,D,DK] -> bf16 Bt[n*D + d] with n = h*DK + dk
__global__ __launch_bounds__(256)
void pack_qkv_kernel(const float* __restrict__ W, __bf16* __restrict__ Bt)
{
    const size_t total = (size_t)D_MODEL * D_MODEL;
    for (size_t i = (size_t)blockIdx.x * blockDim.x + threadIdx.x; i < total;
         i += (size_t)gridDim.x * blockDim.x) {
        int n = (int)(i >> 10);
        int d = (int)(i & 1023);
        int h = n >> 6, dk = n & 63;
        Bt[i] = (__bf16)W[((size_t)h * D_MODEL + d) * HDK + dk];
    }
}

// ---------------------------------------------------------------------------
// WMMA bf16 GEMM v3: C[M,N] = act(A @ B + bias) + residual
//   Block = 8 waves = 32(M) x 512(N) of C.  Wave = 32x64 (2 M-tiles x 4 N-tiles,
//   8 WMMAs per k-step; A-frag reused 4x, B-frag reused 2x).
//   A tile (32x32 bf16, 2KB) staged per k-step into double-buffered LDS via
//   global_load_async_to_lds_b64 (ASYNCcnt), overlapping next tile's DMA with
//   this step's compute.  All 4 B fragments + both A fragments are loaded
//   BEFORE the WMMA burst so the compiler can clause the 8 global_load_b128s
//   and issue 8 back-to-back v_wmma with a single wait.
//   M mult of 32, N mult of 512, K mult of 32 -> EXEC all-1s around WMMA.
// ---------------------------------------------------------------------------
__global__ __launch_bounds__(256)
void gemm_bf16_kernel(const __bf16* __restrict__ A, const __bf16* __restrict__ Bt,
                      const float* __restrict__ bias, const float* __restrict__ resid,
                      float* __restrict__ Cf, __bf16* __restrict__ Cb,
                      int M, int N, int K, int act)
{
    __shared__ __align__(16) char atile[2 * A_TILE_BYTES];

    const int lane   = threadIdx.x & 31;
    const int wave   = threadIdx.x >> 5;
    const int rowbase = blockIdx.y * 32;
    const int tileN0  = blockIdx.x * 32 + wave * 4;
    const int halfs  = (lane < 16) ? 0 : 1;
    const int lrow   = lane & 15;

    // Async stage: thread t copies 8B of the 32x32 A tile (row t/8, chunk t%8)
    const int cp_r = threadIdx.x >> 3;
    const int cp_c = threadIdx.x & 7;
    const unsigned lds_base = (unsigned)(size_t)(void*)atile;
    const __bf16* cp_g0 = A + (size_t)(rowbase + cp_r) * K + cp_c * 4;

    // A layout (matches B): lanes<16 hold K0..15; lanes>=16 hold K16..31
    const int ka_byte = halfs * 32;          // 2 * (halfs*16)
    // B layout (matches A): lanes<16 hold K0..15; lanes>=16 hold K16..31; N = lane&15
    const int kb_off = halfs * 16;

    const __bf16* brow[4];
    for (int j = 0; j < 4; ++j)
        brow[j] = Bt + (size_t)((tileN0 + j) * 16 + lrow) * K + kb_off;

    v8f c0[4] = {v8f{}, v8f{}, v8f{}, v8f{}};   // M-tile 0 (rows 0..15)
    v8f c1[4] = {v8f{}, v8f{}, v8f{}, v8f{}};   // M-tile 1 (rows 16..31)

    const int niter = K >> 5;
    // prologue: stage tile 0
    async_copy_b64(lds_base + cp_r * A_PITCH + cp_c * 8, cp_g0);

    for (int i = 0; i < niter; ++i) {
        const int cur = i & 1;
        const int k0 = i * 32;

        // Issue all 4 B-fragment loads first: 8 global_load_b128 in flight
        // while we wait on the async A copy and read the LDS fragments.
        v16bf b0 = *(const v16bf*)(brow[0] + k0);
        v16bf b1 = *(const v16bf*)(brow[1] + k0);
        v16bf b2 = *(const v16bf*)(brow[2] + k0);
        v16bf b3 = *(const v16bf*)(brow[3] + k0);

        wait_async0();        // my slice of buf[cur] landed
        __syncthreads();      // everyone's slice landed; prev reads of buf[cur^1] done
        if (i + 1 < niter)    // overlap next tile's DMA with this step's compute
            async_copy_b64(lds_base + (cur ^ 1) * A_TILE_BYTES + cp_r * A_PITCH + cp_c * 8,
                           cp_g0 + (size_t)(i + 1) * 32);

        const char* abuf = atile + cur * A_TILE_BYTES;
        v16bf a0, a1;
        {
            v8bf lo = *(const v8bf*)(abuf + lrow * A_PITCH + ka_byte);
            v8bf hi = *(const v8bf*)(abuf + lrow * A_PITCH + ka_byte + 16);
            for (int e = 0; e < 8; ++e) { a0[e] = lo[e]; a0[8 + e] = hi[e]; }
            lo = *(const v8bf*)(abuf + (lrow + 16) * A_PITCH + ka_byte);
            hi = *(const v8bf*)(abuf + (lrow + 16) * A_PITCH + ka_byte + 16);
            for (int e = 0; e < 8; ++e) { a1[e] = lo[e]; a1[8 + e] = hi[e]; }
        }

        // 8 back-to-back WMMAs (B-frag reused across the two M-tiles)
        c0[0] = __builtin_amdgcn_wmma_f32_16x16x32_bf16(false, a0, false, b0, (short)0, c0[0], false, false);
        c1[0] = __builtin_amdgcn_wmma_f32_16x16x32_bf16(false, a1, false, b0, (short)0, c1[0], false, false);
        c0[1] = __builtin_amdgcn_wmma_f32_16x16x32_bf16(false, a0, false, b1, (short)0, c0[1], false, false);
        c1[1] = __builtin_amdgcn_wmma_f32_16x16x32_bf16(false, a1, false, b1, (short)0, c1[1], false, false);
        c0[2] = __builtin_amdgcn_wmma_f32_16x16x32_bf16(false, a0, false, b2, (short)0, c0[2], false, false);
        c1[2] = __builtin_amdgcn_wmma_f32_16x16x32_bf16(false, a1, false, b2, (short)0, c1[2], false, false);
        c0[3] = __builtin_amdgcn_wmma_f32_16x16x32_bf16(false, a0, false, b3, (short)0, c0[3], false, false);
        c1[3] = __builtin_amdgcn_wmma_f32_16x16x32_bf16(false, a1, false, b3, (short)0, c1[3], false, false);
    }

    // Epilogue. C/D layout: VGPR v -> M = v + 8*halfs, N = lane&15.
    for (int mi = 0; mi < 2; ++mi) {
        const v8f* cc = mi ? c1 : c0;
        for (int j = 0; j < 4; ++j) {
            const int n = (tileN0 + j) * 16 + lrow;
            const float bv = bias ? bias[n] : 0.0f;
            for (int v = 0; v < 8; ++v) {
                const int mm = rowbase + mi * 16 + v + halfs * 8;
                const size_t idx = (size_t)mm * N + n;
                float val = cc[j][v] + bv;
                if (act == 1) val = 0.5f * val * (1.0f + erff(val * 0.70710678f));
                if (resid) val += resid[idx];
                if (Cf) Cf[idx] = val;
                if (Cb) Cb[idx] = (__bf16)val;
            }
        }
    }
}

// ---------------------------------------------------------------------------
// Attention for one (b,h): scores -> mask -> softmax -> @V.  ~2% of FLOPs.
// ---------------------------------------------------------------------------
__global__ __launch_bounds__(128)
void attention_kernel(const float* __restrict__ q, const float* __restrict__ k,
                      const float* __restrict__ v, const int* __restrict__ mask,
                      __bf16* __restrict__ o)
{
    const int bh = blockIdx.x;
    const int b = bh / NHEAD, h = bh % NHEAD;
    const int t = threadIdx.x;                 // 0..127
    __shared__ float sc[LSEQ];
    __shared__ float red[LSEQ];
    const float scale = 0.125f;                // 1/sqrt(64)

    const float* kt = k + ((size_t)(b * LSEQ + t) * D_MODEL + h * HDK);
    for (int s = 0; s < LSEQ; ++s) {
        const float* qs = q + ((size_t)(b * LSEQ + s) * D_MODEL + h * HDK);
        float acc = 0.f;
        for (int d = 0; d < HDK; ++d) acc += qs[d] * kt[d];
        acc *= scale;
        if (mask[((size_t)b * LSEQ + s) * LSEQ + t] == 0) acc = -__builtin_inff();
        sc[t] = acc; red[t] = acc; __syncthreads();
        for (int off = 64; off > 0; off >>= 1) {
            if (t < off) red[t] = fmaxf(red[t], red[t + off]);
            __syncthreads();
        }
        const float mx = red[0]; __syncthreads();
        const float e = expf(sc[t] - mx);
        sc[t] = e; red[t] = e; __syncthreads();
        for (int off = 64; off > 0; off >>= 1) {
            if (t < off) red[t] += red[t + off];
            __syncthreads();
        }
        const float inv = 1.0f / red[0]; __syncthreads();
        if (t < HDK) {
            float acc2 = 0.f;
            for (int tt = 0; tt < LSEQ; ++tt)
                acc2 += sc[tt] * v[(size_t)(b * LSEQ + tt) * D_MODEL + h * HDK + t];
            o[(size_t)(b * LSEQ + s) * D_MODEL + h * HDK + t] = (__bf16)(acc2 * inv);
        }
        __syncthreads();
    }
}

// Gather last valid token per batch (idx = sum(mask[b,0,:]) - 1) -> bf16 row
__global__ __launch_bounds__(256)
void select_kernel(const float* __restrict__ x, const int* __restrict__ mask,
                   __bf16* __restrict__ sel)
{
    const int b = blockIdx.x;
    __shared__ int sidx;
    if (threadIdx.x == 0) {
        int c = 0;
        for (int t = 0; t < LSEQ; ++t) c += mask[(size_t)b * LSEQ * LSEQ + t];
        sidx = c - 1;
    }
    __syncthreads();
    const int r = sidx;
    for (int i = threadIdx.x; i < D_MODEL; i += 256)
        sel[b * D_MODEL + i] = (__bf16)x[(size_t)(b * LSEQ + r) * D_MODEL + i];
}

__global__ __launch_bounds__(256)
void l2norm_kernel(float* __restrict__ y)
{
    const int b = blockIdx.x;
    __shared__ float s[256];
    float a = 0.f;
    for (int i = threadIdx.x; i < EMBD; i += 256) { float t = y[b * EMBD + i]; a += t * t; }
    s[threadIdx.x] = a; __syncthreads();
    for (int off = 128; off > 0; off >>= 1) {
        if (threadIdx.x < off) s[threadIdx.x] += s[threadIdx.x + off];
        __syncthreads();
    }
    const float inv = rsqrtf(s[0]);
    for (int i = threadIdx.x; i < EMBD; i += 256) y[b * EMBD + i] *= inv;
}

// ---------------------------------------------------------------------------
extern "C" void kernel_launch(void* const* d_in, const int* in_sizes, int n_in,
                              void* d_out, int out_size, void* d_ws, size_t ws_size,
                              hipStream_t stream)
{
    const int*   text  = (const int*)d_in[0];
    const int*   mask  = (const int*)d_in[1];
    const float* emb   = (const float*)d_in[2];
    const float* cls   = (const float*)d_in[3];
    const float* ln1_g = (const float*)d_in[4];
    const float* ln1_b = (const float*)d_in[5];
    const float* Wq    = (const float*)d_in[6];
    const float* bq    = (const float*)d_in[7];
    const float* Wk    = (const float*)d_in[8];
    const float* bk    = (const float*)d_in[9];
    const float* Wv    = (const float*)d_in[10];
    const float* bv    = (const float*)d_in[11];
    const float* Wo    = (const float*)d_in[12];
    const float* bo    = (const float*)d_in[13];
    const float* ln2_g = (const float*)d_in[14];
    const float* ln2_b = (const float*)d_in[15];
    const float* W1    = (const float*)d_in[16];
    const float* b1    = (const float*)d_in[17];
    const float* W2    = (const float*)d_in[18];
    const float* b2    = (const float*)d_in[19];
    const float* proj  = (const float*)d_in[20];

    // ---- workspace carve (all 256B aligned) ----
    char* p = (char*)d_ws;
    auto carve = [&](size_t bytes) -> void* {
        void* r = (void*)p;
        p += (bytes + 255) & ~(size_t)255;
        return r;
    };
    float*  x    = (float*)  carve((size_t)NTOK * D_MODEL * 4);   // residual stream
    __bf16* hb   = (__bf16*) carve((size_t)NTOK * D_MODEL * 2);   // LN out (A matrix)
    float*  qb   = (float*)  carve((size_t)NTOK * D_MODEL * 4);
    float*  kb   = (float*)  carve((size_t)NTOK * D_MODEL * 4);
    float*  vb   = (float*)  carve((size_t)NTOK * D_MODEL * 4);
    __bf16* ob   = (__bf16*) carve((size_t)NTOK * D_MODEL * 2);   // attn out (A matrix)
    __bf16* ub   = (__bf16*) carve((size_t)NTOK * DFF * 2);       // MLP hidden (A matrix)
    __bf16* selb = (__bf16*) carve((size_t)BATCH * D_MODEL * 2);
    __bf16* wtS  = (__bf16*) carve((size_t)D_MODEL * D_MODEL * 2);// reused Wq/Wk/Wv/Wo
    __bf16* wtB  = (__bf16*) carve((size_t)D_MODEL * DFF * 2);    // reused W1/W2
    __bf16* pjT  = (__bf16*) carve((size_t)D_MODEL * EMBD * 2);
    (void)ws_size; (void)n_in; (void)in_sizes; (void)out_size;

    const dim3 blk(256);
    // GEMM grids: block covers 32(M) x 512(N) of C
    const dim3 gD (D_MODEL / 512, NTOK / 32);   // N=1024
    const dim3 gF (DFF / 512,     NTOK / 32);   // N=4096
    const dim3 gP (1,             BATCH / 32);  // final proj, N=512, M=32

    embed_kernel<<<NTOK, blk, 0, stream>>>(text, emb, cls, x);

    for (int l = 0; l < NLAYER; ++l) {
        const size_t wqkv = (size_t)l * NHEAD * D_MODEL * HDK;
        // ---- attention ----
        layernorm_kernel<<<NTOK, blk, 0, stream>>>(x, ln1_g + l * D_MODEL,
                                                   ln1_b + l * D_MODEL, hb);
        pack_qkv_kernel<<<1024, blk, 0, stream>>>(Wq + wqkv, wtS);
        gemm_bf16_kernel<<<gD, blk, 0, stream>>>(hb, wtS, bq + l * D_MODEL, nullptr,
                                                 qb, nullptr, NTOK, D_MODEL, D_MODEL, 0);
        pack_qkv_kernel<<<1024, blk, 0, stream>>>(Wk + wqkv, wtS);
        gemm_bf16_kernel<<<gD, blk, 0, stream>>>(hb, wtS, bk + l * D_MODEL, nullptr,
                                                 kb, nullptr, NTOK, D_MODEL, D_MODEL, 0);
        pack_qkv_kernel<<<1024, blk, 0, stream>>>(Wv + wqkv, wtS);
        gemm_bf16_kernel<<<gD, blk, 0, stream>>>(hb, wtS, bv + l * D_MODEL, nullptr,
                                                 vb, nullptr, NTOK, D_MODEL, D_MODEL, 0);
        attention_kernel<<<BATCH * NHEAD, dim3(128), 0, stream>>>(qb, kb, vb, mask, ob);
        pack_w_kernel<<<1024, blk, 0, stream>>>(Wo + (size_t)l * D_MODEL * D_MODEL,
                                                wtS, D_MODEL, D_MODEL);
        gemm_bf16_kernel<<<gD, blk, 0, stream>>>(ob, wtS, bo + l * D_MODEL, x,
                                                 x, nullptr, NTOK, D_MODEL, D_MODEL, 0);
        // ---- MLP ----
        layernorm_kernel<<<NTOK, blk, 0, stream>>>(x, ln2_g + l * D_MODEL,
                                                   ln2_b + l * D_MODEL, hb);
        pack_w_kernel<<<2048, blk, 0, stream>>>(W1 + (size_t)l * D_MODEL * DFF,
                                                wtB, D_MODEL, DFF);
        gemm_bf16_kernel<<<gF, blk, 0, stream>>>(hb, wtB, b1 + l * DFF, nullptr,
                                                 nullptr, ub, NTOK, DFF, D_MODEL, 1);
        pack_w_kernel<<<2048, blk, 0, stream>>>(W2 + (size_t)l * DFF * D_MODEL,
                                                wtB, DFF, D_MODEL);
        gemm_bf16_kernel<<<gD, blk, 0, stream>>>(ub, wtB, b2 + l * D_MODEL, x,
                                                 x, nullptr, NTOK, D_MODEL, DFF, 0);
    }

    // ---- head: gather last valid token, project, L2-normalize ----
    select_kernel<<<BATCH, blk, 0, stream>>>(x, mask, selb);
    pack_w_kernel<<<1024, blk, 0, stream>>>(proj, pjT, D_MODEL, EMBD);
    gemm_bf16_kernel<<<gP, blk, 0, stream>>>(selb, pjT, nullptr, nullptr,
                                             (float*)d_out, nullptr,
                                             BATCH, EMBD, D_MODEL, 0);
    l2norm_kernel<<<BATCH, blk, 0, stream>>>((float*)d_out);
}
